// APPNPNodeClassifier_68143951118900
// MI455X (gfx1250) — matrix operand order repaired and checked by reading, not compiled, not run on hardware
//
#include <hip/hip_runtime.h>
#include <stdint.h>
#include <stddef.h>

#define NN    50000
#define NE    800000
#define INC   512
#define HIDC  256
#define OUTC  64
#define KPROP 10
#define ALPHA 0.1f

typedef __attribute__((ext_vector_type(16))) __bf16 v16bf;
typedef __attribute__((ext_vector_type(8)))  __bf16 v8bf;
typedef __attribute__((ext_vector_type(8)))  float  v8f;

// ---------------------------------------------------------------- setup ----

__global__ void zero_u32(unsigned* __restrict__ p, int n) {
    int i = blockIdx.x * 256 + threadIdx.x;
    if (i < n) p[i] = 0u;
}

// edge_index is int64 [2, E] row-major: row0 = src, row1 = dst
__global__ void edges_prep(const long long* __restrict__ ei, int* __restrict__ src32,
                           int* __restrict__ dst32, unsigned* __restrict__ deg) {
    int i = blockIdx.x * 256 + threadIdx.x;
    if (i >= NE) return;
    int s = (int)ei[i];
    int d = (int)ei[NE + i];
    src32[i] = s;
    dst32[i] = d;
    atomicAdd(&deg[d], 1u);
}

__global__ void dinv_kernel(const unsigned* __restrict__ deg, float* __restrict__ dinv,
                            float* __restrict__ self_nrm) {
    int i = blockIdx.x * 256 + threadIdx.x;
    if (i >= NN) return;
    float dv = rsqrtf((float)deg[i] + 1.0f);
    dinv[i] = dv;
    self_nrm[i] = dv * dv;
}

// ------------------------------------------------ exclusive scan (rowptr) --

__global__ void scan_local(const unsigned* __restrict__ deg, int* __restrict__ lexcl,
                           unsigned* __restrict__ bsum) {
    __shared__ unsigned sh[256];
    int t = threadIdx.x;
    int idx = blockIdx.x * 256 + t;
    unsigned v = (idx < NN) ? deg[idx] : 0u;
    sh[t] = v;
    __syncthreads();
#pragma unroll
    for (int o = 1; o < 256; o <<= 1) {
        unsigned add = (t >= o) ? sh[t - o] : 0u;
        __syncthreads();
        sh[t] += add;
        __syncthreads();
    }
    if (idx < NN) lexcl[idx] = (int)(sh[t] - v);
    if (t == 255) bsum[blockIdx.x] = sh[255];
}

__global__ void scan_sums(unsigned* __restrict__ bsum, int nb) {
    __shared__ unsigned sh[256];
    int t = threadIdx.x;
    unsigned v = (t < nb) ? bsum[t] : 0u;
    sh[t] = v;
    __syncthreads();
#pragma unroll
    for (int o = 1; o < 256; o <<= 1) {
        unsigned add = (t >= o) ? sh[t - o] : 0u;
        __syncthreads();
        sh[t] += add;
        __syncthreads();
    }
    bsum[t] = sh[t] - v;   // exclusive
}

__global__ void scan_add(int* __restrict__ rowptr, const unsigned* __restrict__ bsum,
                         int* __restrict__ cursor) {
    int idx = blockIdx.x * 256 + threadIdx.x;
    if (idx < NN) {
        int r = rowptr[idx] + (int)bsum[idx >> 8];
        rowptr[idx] = r;
        cursor[idx] = r;
    }
    if (idx == 0) rowptr[NN] = NE;
}

__global__ void fill_csr(const int* __restrict__ src32, const int* __restrict__ dst32,
                         const float* __restrict__ dinv, int* __restrict__ cursor,
                         int* __restrict__ csr_src, float* __restrict__ csr_w) {
    int e = blockIdx.x * 256 + threadIdx.x;
    if (e >= NE) return;
    int s = src32[e], d = dst32[e];
    int pos = atomicAdd(&cursor[d], 1);
    csr_src[pos] = s;
    csr_w[pos] = dinv[s] * dinv[d];
}

__global__ void cvt_bf16(const float* __restrict__ in, __bf16* __restrict__ out, int n) {
    int i = blockIdx.x * 256 + threadIdx.x;
    if (i < n) out[i] = (__bf16)in[i];
}

// ------------------------------------------------------------- WMMA GEMM --
// out[M,Nout] = act(A[M,KD] @ W[Nout,KD]^T + bias).  One 16x16 tile per wave.
// Fragment packing per CDNA5 ISA 7.12.2:
//   A 16x32 bf16: lane half h, element j -> k = j + 8*h + (j>=8 ? 8 : 0)
//   B 32x16 bf16: lane half h, element j -> k = j + 8*h + (j>=8 ? 8 : 0)   (same as A)
//   C/D f32:      lane l, vgpr v -> (M = v + 8*(l>>4), N = l&15)

template <bool A_BF16, bool OUT_BF16, bool RELU>
__global__ void wmma_gemm(const void* __restrict__ Ap, const __bf16* __restrict__ Wb,
                          const float* __restrict__ bias, void* __restrict__ outp,
                          int KD, int NT, int total_tiles) {
    int lane = threadIdx.x & 31;
    int wv   = threadIdx.x >> 5;
    int tile = blockIdx.x * 8 + wv;
    if (tile >= total_tiles) return;            // wave-uniform: EXEC stays all-1

    int mrow = (tile / NT) * 16;
    int ncol = (tile % NT) * 16;
    int half = lane >> 4;
    int l16  = lane & 15;
    int mr   = mrow + l16;
    int nr   = ncol + l16;
    int Nout = NT * 16;

    v8f c = {};
    const __bf16* wr = Wb + (size_t)nr * KD;

    for (int k0 = 0; k0 < KD; k0 += 32) {
        v16bf a;
        int kA1 = k0 + half * 8;
        int kA2 = k0 + 16 + half * 8;
        if constexpr (A_BF16) {
            const __bf16* ar = (const __bf16*)Ap + (size_t)mr * KD;
            v8bf r0 = *(const v8bf*)(ar + kA1);
            v8bf r1 = *(const v8bf*)(ar + kA2);
#pragma unroll
            for (int j = 0; j < 8; ++j) { a[j] = r0[j]; a[8 + j] = r1[j]; }
        } else {
            const float* ar = (const float*)Ap + (size_t)mr * KD;
            float4 f0 = *(const float4*)(ar + kA1);
            float4 f1 = *(const float4*)(ar + kA1 + 4);
            float4 f2 = *(const float4*)(ar + kA2);
            float4 f3 = *(const float4*)(ar + kA2 + 4);
            a[0]  = (__bf16)f0.x; a[1]  = (__bf16)f0.y; a[2]  = (__bf16)f0.z; a[3]  = (__bf16)f0.w;
            a[4]  = (__bf16)f1.x; a[5]  = (__bf16)f1.y; a[6]  = (__bf16)f1.z; a[7]  = (__bf16)f1.w;
            a[8]  = (__bf16)f2.x; a[9]  = (__bf16)f2.y; a[10] = (__bf16)f2.z; a[11] = (__bf16)f2.w;
            a[12] = (__bf16)f3.x; a[13] = (__bf16)f3.y; a[14] = (__bf16)f3.z; a[15] = (__bf16)f3.w;
        }
        v8bf b0 = *(const v8bf*)(wr + k0 + half * 8);   // 16B aligned runs, table order
        v8bf b1 = *(const v8bf*)(wr + k0 + 16 + half * 8);
        v16bf b;
#pragma unroll
        for (int j = 0; j < 8; ++j) { b[j] = b0[j]; b[8 + j] = b1[j]; }
        c = __builtin_amdgcn_wmma_f32_16x16x32_bf16(false, a, false, b,
                                                    (short)0, c, false, false);
    }

    float bs = bias[nr];
#pragma unroll
    for (int v = 0; v < 8; ++v) {
        int m = mrow + v + half * 8;
        float val = c[v] + bs;
        if (RELU) val = val > 0.f ? val : 0.f;
        if constexpr (OUT_BF16)
            ((__bf16*)outp)[(size_t)m * Nout + nr] = (__bf16)val;
        else
            ((float*)outp)[(size_t)m * Nout + nr] = val;
    }
}

// ------------------------------------------------------- APPNP propagation -
// One wave per node; 32 lanes x float2 cover the 64-wide feature row.
// Pure gather over incoming-edge CSR (no atomics); h state lives in L2.

__global__ void appnp_step(const float* __restrict__ hcur, const float* __restrict__ z,
                           const int* __restrict__ rowptr, const int* __restrict__ csr_src,
                           const float* __restrict__ csr_w, const float* __restrict__ self_nrm,
                           float* __restrict__ hout) {
    int gid  = blockIdx.x * blockDim.x + threadIdx.x;
    int node = gid >> 5;
    int lane = threadIdx.x & 31;
    if (node >= NN) return;

    int beg = rowptr[node], end = rowptr[node + 1];
    float accx = 0.f, accy = 0.f;
    for (int e = beg; e < end; ++e) {
        int   s   = __builtin_amdgcn_readfirstlane(csr_src[e]);
        float wgt = csr_w[e];
        if (e + 1 < end)
            __builtin_prefetch(hcur + (size_t)csr_src[e + 1] * OUTC, 0, 1);
        float2 hv = *(const float2*)(hcur + (size_t)s * OUTC + lane * 2);
        accx = fmaf(wgt, hv.x, accx);
        accy = fmaf(wgt, hv.y, accy);
    }
    float  sn = self_nrm[node];
    float2 hs = *(const float2*)(hcur + (size_t)node * OUTC + lane * 2);
    float2 zv = *(const float2*)(z    + (size_t)node * OUTC + lane * 2);
    float2 o;
    o.x = (1.f - ALPHA) * (accx + sn * hs.x) + ALPHA * zv.x;
    o.y = (1.f - ALPHA) * (accy + sn * hs.y) + ALPHA * zv.y;
    *(float2*)(hout + (size_t)node * OUTC + lane * 2) = o;
}

// ------------------------------------------------------------------ launch -

extern "C" void kernel_launch(void* const* d_in, const int* in_sizes, int n_in,
                              void* d_out, int out_size, void* d_ws, size_t ws_size,
                              hipStream_t stream) {
    (void)in_sizes; (void)n_in; (void)out_size; (void)ws_size;

    const float*     x  = (const float*)d_in[0];
    const long long* ei = (const long long*)d_in[1];
    const float*     W1 = (const float*)d_in[2];
    const float*     b1 = (const float*)d_in[3];
    const float*     W2 = (const float*)d_in[4];
    const float*     b2 = (const float*)d_in[5];

    char* base = (char*)d_ws;
    size_t off = 0;
    auto alloc = [&](size_t bytes) -> void* {
        void* p = base + off;
        off += (bytes + 255) & ~(size_t)255;
        return p;
    };

    unsigned* deg      = (unsigned*)alloc((size_t)NN * 4);
    float*    dinv     = (float*)   alloc((size_t)NN * 4);
    float*    self_nrm = (float*)   alloc((size_t)NN * 4);
    int*      src32    = (int*)     alloc((size_t)NE * 4);
    int*      dst32    = (int*)     alloc((size_t)NE * 4);
    int*      rowptr   = (int*)     alloc((size_t)(NN + 1) * 4);
    int*      cursor   = (int*)     alloc((size_t)NN * 4);
    unsigned* bsum     = (unsigned*)alloc(256 * 4);
    int*      csr_src  = (int*)     alloc((size_t)NE * 4);
    float*    csr_w    = (float*)   alloc((size_t)NE * 4);
    __bf16*   W1b      = (__bf16*)  alloc((size_t)INC * HIDC * 2);
    __bf16*   W2b      = (__bf16*)  alloc((size_t)HIDC * OUTC * 2);
    __bf16*   hbuf     = (__bf16*)  alloc((size_t)NN * HIDC * 2);
    float*    zbuf     = (float*)   alloc((size_t)NN * OUTC * 4);
    float*    hA       = (float*)   alloc((size_t)NN * OUTC * 4);
    float*    hB       = (float*)   alloc((size_t)NN * OUTC * 4);

    dim3 B(256);
    int nbN = (NN + 255) / 256;          // 196
    int nbE = (NE + 255) / 256;

    zero_u32  <<<nbN, B, 0, stream>>>(deg, NN);
    edges_prep<<<nbE, B, 0, stream>>>(ei, src32, dst32, deg);
    dinv_kernel<<<nbN, B, 0, stream>>>(deg, dinv, self_nrm);
    scan_local<<<nbN, B, 0, stream>>>(deg, rowptr, bsum);
    scan_sums <<<1,   B, 0, stream>>>(bsum, nbN);
    scan_add  <<<nbN, B, 0, stream>>>(rowptr, bsum, cursor);
    fill_csr  <<<nbE, B, 0, stream>>>(src32, dst32, dinv, cursor, csr_src, csr_w);
    cvt_bf16  <<<(INC * HIDC + 255) / 256, B, 0, stream>>>(W1, W1b, INC * HIDC);
    cvt_bf16  <<<(HIDC * OUTC + 255) / 256, B, 0, stream>>>(W2, W2b, HIDC * OUTC);

    // GEMM1: h = relu(x @ W1^T + b1)   (f32 A -> bf16 out)
    int tiles1 = (NN / 16) * (HIDC / 16);        // 3125 * 16 = 50000 waves
    wmma_gemm<false, true, true><<<(tiles1 + 7) / 8, B, 0, stream>>>(
        x, W1b, b1, hbuf, INC, HIDC / 16, tiles1);

    // GEMM2: z = h @ W2^T + b2         (bf16 A -> f32 out)
    int tiles2 = (NN / 16) * (OUTC / 16);        // 3125 * 4 = 12500 waves
    wmma_gemm<true, false, false><<<(tiles2 + 7) / 8, B, 0, stream>>>(
        hbuf, W2b, b2, zbuf, HIDC, OUTC / 16, tiles2);

    // K propagation steps: hnew = 0.9*(CSR-gather + self) + 0.1*z
    const float* cur = zbuf;
    float* pp[2] = {hA, hB};
    int nwb = (NN * 32 + 255) / 256;             // one wave per node -> 6250 blocks
    for (int k = 0; k < KPROP; ++k) {
        float* o = (k == KPROP - 1) ? (float*)d_out : pp[k & 1];
        appnp_step<<<nwb, B, 0, stream>>>(cur, zbuf, rowptr, csr_src, csr_w, self_nrm, o);
        cur = o;
    }
}
